// ParallelTransformerBlock_19516331393048
// MI455X (gfx1250) — matrix operand order repaired and checked by reading, not compiled, not run on hardware
//
#include <hip/hip_runtime.h>
#include <hip/hip_bf16.h>

#define DIM 2048
#define HEADS 16
#define DH 128
#define FF_INNER 8192
#define ATTN_INNER 2048
#define FUSED_TOTAL 18688   // 2048 + 128 + 128 + 16384
#define QKV_COLS 2304       // 2048 + 128 + 128
#define BATCH 2
#define SEQ 2048
#define NTOK (BATCH * SEQ)  // 4096
#define RANK 8

typedef __bf16 v16bf __attribute__((ext_vector_type(16)));
typedef float  v8f   __attribute__((ext_vector_type(8)));

union Frag {
  v16bf v;
  uint4 q[2];
  unsigned short u[16];
};

// ---- CDNA5 async / TDM availability (device pass only; host pass -> fallbacks)
#if __has_builtin(__builtin_amdgcn_global_load_async_to_lds_b128) && \
    __has_builtin(__builtin_amdgcn_s_wait_asynccnt)
#define HAS_ASYNC_LDS 1
#else
#define HAS_ASYNC_LDS 0
#endif
#if __has_builtin(__builtin_amdgcn_tensor_load_to_lds) && \
    __has_builtin(__builtin_amdgcn_s_wait_tensorcnt)
#define HAS_TDM 1
#else
#define HAS_TDM 0
#endif

// gcc-style int4 vector: matches the builtin's parameter type
// ('__attribute__((__vector_size__(4*sizeof(int)))) int *', generic AS).
typedef int gv4i __attribute__((vector_size(16)));
typedef __attribute__((address_space(3))) gv4i* lds_v4ip;

// copy 16 bytes global -> LDS (async when available)
__device__ __forceinline__ void copy16_g2l(void* lds_dst, const void* gsrc) {
#if HAS_ASYNC_LDS
  __builtin_amdgcn_global_load_async_to_lds_b128((gv4i*)gsrc, (lds_v4ip)lds_dst, 0, 0);
#else
  *(uint4*)lds_dst = *(const uint4*)gsrc;
#endif
}
__device__ __forceinline__ void async_fence() {
#if HAS_ASYNC_LDS
  __builtin_amdgcn_s_wait_asynccnt(0);
#endif
}

#if HAS_TDM
typedef unsigned int u32x4 __attribute__((ext_vector_type(4)));
typedef int i32x8 __attribute__((ext_vector_type(8)));
typedef int i32x4 __attribute__((ext_vector_type(4)));
// 2D tile load via Tensor Data Mover (6-arg builtin form).
// tile_w/tile_h in elements (2B each), row_stride in elements.
// pad: 16B appended in LDS after every 256B row (-> 136-ushort LDS stride).
__device__ __forceinline__ void tdm_load_2d(unsigned lds_off, const void* gaddr,
                                            unsigned tile_w, unsigned tile_h,
                                            unsigned row_stride,
                                            unsigned pad_en, unsigned pad_interval_code,
                                            unsigned pad_amount_code) {
  unsigned long long ga = (unsigned long long)(size_t)gaddr;
  u32x4 g0;
  g0.x = 1u;                                                 // count=1, user mode
  g0.y = lds_off;                                            // lds_addr (bytes)
  g0.z = (unsigned)(ga & 0xffffffffu);                       // global_addr[31:0]
  g0.w = (unsigned)((ga >> 32) & 0x01ffffffu) | 0x80000000u; // addr[56:32] | type=2
  unsigned td0 = row_stride;      // tensor dim0 (elems)
  unsigned td1 = 1u << 20;        // tensor dim1: large (tiles always in-bounds)
  i32x8 g1;
  g1[0] = (int)((1u << 16) | (pad_en << 20) |
                (pad_interval_code << 22) | (pad_amount_code << 25)); // data_size=2B
  g1[1] = (int)((td0 & 0xffffu) << 16);                    // tensor_dim0[15:0]
  g1[2] = (int)((td0 >> 16) | ((td1 & 0xffffu) << 16));    // td0[31:16] | td1[15:0]
  g1[3] = (int)((td1 >> 16) | (tile_w << 16));             // td1[31:16] | tile_dim0
  g1[4] = (int)(tile_h & 0xffffu);                         // tile_dim1 (tile_dim2=0)
  g1[5] = (int)row_stride;                                 // tensor_dim0_stride[31:0]
  g1[6] = 0;                                               // stride0 hi | stride1 lo
  g1[7] = 0;
  i32x4 z4 = {0, 0, 0, 0};
  i32x8 z8 = {0, 0, 0, 0, 0, 0, 0, 0};
  __builtin_amdgcn_tensor_load_to_lds(g0, g1, z4, z4, z8, 0);
}
#endif

__device__ __forceinline__ unsigned short f32_to_bf16(float f) {
  unsigned int u = __float_as_uint(f);
  u = (u + 0x7fffu + ((u >> 16) & 1u)) >> 16;
  return (unsigned short)u;
}
__device__ __forceinline__ float bf16_to_f32(unsigned short s) {
  return __uint_as_float(((unsigned int)s) << 16);
}

// ---------------------------------------------------------------- LayerNorm
__global__ __launch_bounds__(256)
void layernorm_kernel(const float* __restrict__ x, const float* __restrict__ gamma,
                      unsigned short* __restrict__ xn) {
  int row = blockIdx.x;
  int tid = threadIdx.x;
  __shared__ float red[256];
  const float* xr = x + (size_t)row * DIM;
  float vals[8];
  float s = 0.f, s2 = 0.f;
  for (int i = 0; i < 8; ++i) {
    float v = xr[tid + i * 256];
    vals[i] = v; s += v; s2 += v * v;
  }
  red[tid] = s; __syncthreads();
  for (int o = 128; o > 0; o >>= 1) { if (tid < o) red[tid] += red[tid + o]; __syncthreads(); }
  float mean = red[0] * (1.0f / DIM);
  __syncthreads();
  red[tid] = s2; __syncthreads();
  for (int o = 128; o > 0; o >>= 1) { if (tid < o) red[tid] += red[tid + o]; __syncthreads(); }
  float var = red[0] * (1.0f / DIM) - mean * mean;
  float rstd = rsqrtf(var + 1e-5f);
  for (int i = 0; i < 8; ++i) {
    int c = tid + i * 256;
    xn[(size_t)row * DIM + c] = f32_to_bf16((vals[i] - mean) * rstd * gamma[c]);
  }
}

// ---------------------------------------------------------------- f32 -> bf16
__global__ __launch_bounds__(256)
void cvt_bf16_kernel(const float* __restrict__ src, unsigned short* __restrict__ dst, int n) {
  int stride = gridDim.x * blockDim.x;
  for (int i = blockIdx.x * blockDim.x + threadIdx.x; i < n; i += stride)
    dst[i] = f32_to_bf16(src[i]);
}

// ---------------------------------------------------------------- bf16 GEMM (C = A@B or C += A@B)
__global__ __launch_bounds__(256)
void gemm_bf16_kernel(const unsigned short* __restrict__ A, int lda,
                      const unsigned short* __restrict__ Bw, int ldb,
                      float* __restrict__ C, int ldc, int K, int accumulate) {
  __shared__ unsigned short sA[128 * 32];
  __shared__ unsigned short sB[128 * 40];  // transposed [n][k], padded
  int tid = threadIdx.x;
  int lane = tid & 31, wave = tid >> 5;
  int hi = lane >> 4, l15 = lane & 15;
  int wm = wave >> 2, wn = wave & 3;
  int m0 = blockIdx.x * 128, n0 = blockIdx.y * 128;

  v8f zero = {0.f, 0.f, 0.f, 0.f, 0.f, 0.f, 0.f, 0.f};
  v8f acc[4][2];
  for (int i = 0; i < 4; ++i) for (int j = 0; j < 2; ++j) acc[i][j] = zero;

  int nk = K >> 5;
  for (int kk = 0; kk < nk; ++kk) {
    int k0 = kk << 5;
    __syncthreads();
    // A tile: 512 x 16B chunks, async global->LDS
    for (int p = 0; p < 2; ++p) {
      int c = tid + p * 256;
      int r = c >> 2, w = c & 3;
      copy16_g2l(&sA[r * 32 + w * 8], &A[(size_t)(m0 + r) * lda + k0 + w * 8]);
    }
    // B tile transposed (scalar; transpose not expressible as bulk copy)
    for (int e = tid; e < 32 * 128; e += 256) {
      int k = e >> 7, n = e & 127;
      sB[n * 40 + k] = Bw[(size_t)(k0 + k) * ldb + n0 + n];
    }
    if (kk + 1 < nk) {
      __builtin_prefetch(&Bw[(size_t)(k0 + 32) * ldb + n0 + (tid & 127)], 0, 1);
      __builtin_prefetch(&A[(size_t)(m0 + (tid >> 1)) * lda + k0 + 32], 0, 1);
    }
    async_fence();
    __syncthreads();

    Frag bfr[2];
    for (int ni = 0; ni < 2; ++ni) {
      int n = wn * 32 + ni * 16 + l15;
      bfr[ni].q[0] = *(const uint4*)&sB[n * 40 + hi * 8];
      bfr[ni].q[1] = *(const uint4*)&sB[n * 40 + hi * 8 + 16];
    }
    for (int mi = 0; mi < 4; ++mi) {
      Frag afr;
      int m = wm * 64 + mi * 16 + l15;
      afr.q[0] = *(const uint4*)&sA[m * 32 + hi * 8];
      afr.q[1] = *(const uint4*)&sA[m * 32 + hi * 8 + 16];
      acc[mi][0] = __builtin_amdgcn_wmma_f32_16x16x32_bf16(
          false, afr.v, false, bfr[0].v, (short)0, acc[mi][0], false, false);
      acc[mi][1] = __builtin_amdgcn_wmma_f32_16x16x32_bf16(
          false, afr.v, false, bfr[1].v, (short)0, acc[mi][1], false, false);
    }
  }
  for (int mi = 0; mi < 4; ++mi)
    for (int ni = 0; ni < 2; ++ni) {
      int col = n0 + wn * 32 + ni * 16 + l15;
      int rowb = m0 + wm * 64 + mi * 16 + hi * 8;
      for (int r = 0; r < 8; ++r) {
        size_t idx = (size_t)(rowb + r) * ldc + col;
        if (accumulate) C[idx] += acc[mi][ni][r];
        else            C[idx]  = acc[mi][ni][r];
      }
    }
}

// ---------------------------------------------------------------- FF GEMM, paired x1/gate + SiLU
__global__ __launch_bounds__(256)
void gemm_ff_silu_kernel(const unsigned short* __restrict__ A,
                         const unsigned short* __restrict__ Bw,
                         unsigned short* __restrict__ Hout) {
  __shared__ unsigned short sA[128 * 32];
  __shared__ unsigned short sB1[128 * 40];
  __shared__ unsigned short sB2[128 * 40];
  int tid = threadIdx.x;
  int lane = tid & 31, wave = tid >> 5;
  int hi = lane >> 4, l15 = lane & 15;
  int wm = wave >> 2, wn = wave & 3;
  int m0 = blockIdx.x * 128, n0 = blockIdx.y * 128;
  const unsigned short* B1 = Bw + QKV_COLS + n0;              // x1 columns
  const unsigned short* B2 = Bw + QKV_COLS + FF_INNER + n0;   // gate columns

  v8f zero = {0.f, 0.f, 0.f, 0.f, 0.f, 0.f, 0.f, 0.f};
  v8f acc1[4][2], acc2[4][2];
  for (int i = 0; i < 4; ++i) for (int j = 0; j < 2; ++j) { acc1[i][j] = zero; acc2[i][j] = zero; }

  for (int kk = 0; kk < (DIM >> 5); ++kk) {
    int k0 = kk << 5;
    __syncthreads();
    for (int p = 0; p < 2; ++p) {
      int c = tid + p * 256;
      int r = c >> 2, w = c & 3;
      copy16_g2l(&sA[r * 32 + w * 8], &A[(size_t)(m0 + r) * DIM + k0 + w * 8]);
    }
    for (int e = tid; e < 32 * 128; e += 256) {
      int k = e >> 7, n = e & 127;
      sB1[n * 40 + k] = B1[(size_t)(k0 + k) * FUSED_TOTAL + n];
      sB2[n * 40 + k] = B2[(size_t)(k0 + k) * FUSED_TOTAL + n];
    }
    async_fence();
    __syncthreads();

    Frag b1[2], b2[2];
    for (int ni = 0; ni < 2; ++ni) {
      int n = wn * 32 + ni * 16 + l15;
      b1[ni].q[0] = *(const uint4*)&sB1[n * 40 + hi * 8];
      b1[ni].q[1] = *(const uint4*)&sB1[n * 40 + hi * 8 + 16];
      b2[ni].q[0] = *(const uint4*)&sB2[n * 40 + hi * 8];
      b2[ni].q[1] = *(const uint4*)&sB2[n * 40 + hi * 8 + 16];
    }
    for (int mi = 0; mi < 4; ++mi) {
      Frag afr;
      int m = wm * 64 + mi * 16 + l15;
      afr.q[0] = *(const uint4*)&sA[m * 32 + hi * 8];
      afr.q[1] = *(const uint4*)&sA[m * 32 + hi * 8 + 16];
      for (int ni = 0; ni < 2; ++ni) {
        acc1[mi][ni] = __builtin_amdgcn_wmma_f32_16x16x32_bf16(
            false, afr.v, false, b1[ni].v, (short)0, acc1[mi][ni], false, false);
        acc2[mi][ni] = __builtin_amdgcn_wmma_f32_16x16x32_bf16(
            false, afr.v, false, b2[ni].v, (short)0, acc2[mi][ni], false, false);
      }
    }
  }
  for (int mi = 0; mi < 4; ++mi)
    for (int ni = 0; ni < 2; ++ni) {
      int col = n0 + wn * 32 + ni * 16 + l15;
      int rowb = m0 + wm * 64 + mi * 16 + hi * 8;
      for (int r = 0; r < 8; ++r) {
        float x1 = acc1[mi][ni][r];
        float g  = acc2[mi][ni][r];
        float h  = x1 * g * (1.0f / (1.0f + __expf(-g)));
        Hout[(size_t)(rowb + r) * FF_INNER + col] = f32_to_bf16(h);
      }
    }
}

// ---------------------------------------------------------------- LoRA: T = A_bf16 @ a (K=2048, R=8)
__global__ __launch_bounds__(256)
void lora_t_kernel(const unsigned short* __restrict__ A, const float* __restrict__ amat,
                   float* __restrict__ T) {
  int tok = blockIdx.x;
  int tid = threadIdx.x;
  int c = tid & 7, chunk = tid >> 3;
  const unsigned short* ar = A + (size_t)tok * DIM;
  float s = 0.f;
  for (int k = chunk * 64; k < chunk * 64 + 64; ++k)
    s += bf16_to_f32(ar[k]) * amat[k * RANK + c];
  __shared__ float red[256];
  red[tid] = s; __syncthreads();
  for (int off = 128; off >= 8; off >>= 1) { if (tid < off) red[tid] += red[tid + off]; __syncthreads(); }
  if (tid < 8) T[(size_t)tok * RANK + tid] = red[tid];
}

// ---------------------------------------------------------------- LoRA apply
__global__ __launch_bounds__(256)
void lora_apply_kernel(const float* __restrict__ T, const float* __restrict__ bm,
                       float* __restrict__ dst, int cols, int ldd, int col0) {
  int tok = blockIdx.x;
  float t[RANK];
  for (int r = 0; r < RANK; ++r) t[r] = T[(size_t)tok * RANK + r];
  for (int c = threadIdx.x; c < cols; c += 256) {
    float s = 0.f;
    for (int r = 0; r < RANK; ++r) s += t[r] * bm[r * cols + c];
    dst[(size_t)tok * ldd + col0 + c] += s;
  }
}

// ---------------------------------------------------------------- RoPE + scale + split to bf16 Q/K/V
__global__ __launch_bounds__(256)
void rope_qkv_kernel(const float* __restrict__ qkv,
                     unsigned short* __restrict__ qg,
                     unsigned short* __restrict__ kg,
                     unsigned short* __restrict__ vg) {
  int tok = blockIdx.x;
  int b = tok >> 11, n = tok & 2047;
  const float* row = qkv + (size_t)tok * QKV_COLS;
  const float qscale = 0.08838834764831845f;      // 128^-0.5
  const float lnf = 9.210340371976184f / 64.0f;   // ln(10000)/64
  for (int i = threadIdx.x; i < ATTN_INNER; i += 256) {
    int h = i >> 7, d = i & 127;
    float ang = (float)n * __expf(-(float)(d & 63) * lnf);
    float t = row[i] * qscale;
    float other = (d < 64) ? -row[h * 128 + d + 64] * qscale
                           :  row[h * 128 + d - 64] * qscale;
    float val = t * __cosf(ang) + other * __sinf(ang);
    qg[((size_t)(b * HEADS + h) * SEQ + n) * DH + d] = f32_to_bf16(val);
  }
  for (int d = threadIdx.x; d < DH; d += 256) {
    float ang = (float)n * __expf(-(float)(d & 63) * lnf);
    float t = row[ATTN_INNER + d];
    float other = (d < 64) ? -row[ATTN_INNER + d + 64] : row[ATTN_INNER + d - 64];
    kg[((size_t)b * SEQ + n) * DH + d] = f32_to_bf16(t * __cosf(ang) + other * __sinf(ang));
    vg[((size_t)b * SEQ + n) * DH + d] = f32_to_bf16(row[ATTN_INNER + DH + d]);
  }
}

// ---------------------------------------------------------------- Flash attention (MQA, causal)
__global__ __launch_bounds__(256)
void flash_attn_kernel(const unsigned short* __restrict__ qg,
                       const unsigned short* __restrict__ kg,
                       const unsigned short* __restrict__ vg,
                       unsigned short* __restrict__ outg) {
  __shared__ unsigned short sK[64 * 136];    // [key][dim], 16B pad per 256B row
  __shared__ unsigned short sVt[128 * 72];   // [dim][key]
  __shared__ unsigned short sP[8 * 16 * 72]; // per-wave P tile [16][72]
  int bh = blockIdx.x, qi = blockIdx.y;
  int b = bh >> 4, h = bh & 15;
  int tid = threadIdx.x;
  int lane = tid & 31, wave = tid >> 5;
  int hi = lane >> 4, l15 = lane & 15;

  // Q fragments in registers (A operand: lane = query)
  Frag qf[4];
  int qrow = qi * 128 + wave * 16 + l15;
  const unsigned short* qbase = qg + ((size_t)bh * SEQ + qrow) * DH;
  for (int c = 0; c < 4; ++c) {
    qf[c].q[0] = *(const uint4*)&qbase[c * 32 + hi * 8];
    qf[c].q[1] = *(const uint4*)&qbase[c * 32 + hi * 8 + 16];
  }

  v8f zero = {0.f, 0.f, 0.f, 0.f, 0.f, 0.f, 0.f, 0.f};
  v8f o[8];
  for (int t = 0; t < 8; ++t) o[t] = zero;
  float m_i[8], l_i[8];
  for (int r = 0; r < 8; ++r) { m_i[r] = -3.0e38f; l_i[r] = 0.f; }

#if HAS_TDM
  unsigned sK_off = (unsigned)(unsigned long long)(__attribute__((address_space(3))) unsigned short*)sK;
#endif

  int nblk = 2 * qi + 2;  // 64-key blocks up to diagonal
  for (int j = 0; j < nblk; ++j) {
    __syncthreads();
    const unsigned short* kgsrc = kg + ((size_t)b * SEQ + j * 64) * DH;
#if HAS_TDM
    // K tile [64][128] via Tensor Data Mover (wave 0 issues one descriptor)
    if (wave == 0) {
      tdm_load_2d(sK_off, kgsrc, /*tile_w=*/DH, /*tile_h=*/64, /*row_stride=*/DH,
                  /*pad_en=*/1, /*pad_interval=64 DWORDs*/5, /*pad_amount=4 DWORDs*/3);
    }
#else
    for (int c = tid; c < 64 * 16; c += 256) {
      int row = c >> 4, w = c & 15;
      copy16_g2l(&sK[row * 136 + w * 8], &kgsrc[(size_t)row * DH + w * 8]);
    }
#endif
    // V tile transposed -> sVt[dim][key] (scalar transpose)
    for (int e = tid; e < 64 * 128; e += 256) {
      int key = e >> 7, d = e & 127;
      sVt[d * 72 + key] = vg[((size_t)b * SEQ + j * 64 + key) * DH + d];
    }
#if HAS_TDM
    if (wave == 0) __builtin_amdgcn_s_wait_tensorcnt(0);
#else
    async_fence();
#endif
    __syncthreads();

    // S = Q K^T : 4 key-subtiles x 4 dim-chunks
    v8f s[4];
    for (int t = 0; t < 4; ++t) s[t] = zero;
    for (int t = 0; t < 4; ++t) {
      int n = t * 16 + l15;
      for (int c = 0; c < 4; ++c) {
        Frag kf;
        kf.q[0] = *(const uint4*)&sK[n * 136 + c * 32 + hi * 8];
        kf.q[1] = *(const uint4*)&sK[n * 136 + c * 32 + hi * 8 + 16];
        s[t] = __builtin_amdgcn_wmma_f32_16x16x32_bf16(
            false, qf[c].v, false, kf.v, (short)0, s[t], false, false);
      }
    }
    if (j >= 2 * qi) {  // causal mask near diagonal
      for (int t = 0; t < 4; ++t)
        for (int r = 0; r < 8; ++r) {
          int col = j * 64 + t * 16 + l15;
          int row = qi * 128 + wave * 16 + hi * 8 + r;
          if (col > row) s[t][r] = -3.0e38f;
        }
    }
    // online softmax per row
    for (int r = 0; r < 8; ++r) {
      float mx = s[0][r];
      for (int t = 1; t < 4; ++t) mx = fmaxf(mx, s[t][r]);
      for (int d = 1; d < 16; d <<= 1) mx = fmaxf(mx, __shfl_xor(mx, d, 32));
      float newm = fmaxf(m_i[r], mx);
      float alpha = __expf(m_i[r] - newm);
      float psum = 0.f;
      for (int t = 0; t < 4; ++t) {
        float p = __expf(s[t][r] - newm);
        s[t][r] = p;
        psum += p;
      }
      for (int d = 1; d < 16; d <<= 1) psum += __shfl_xor(psum, d, 32);
      l_i[r] = l_i[r] * alpha + psum;
      m_i[r] = newm;
      for (int t2 = 0; t2 < 8; ++t2) o[t2][r] *= alpha;
    }
    // stage P through per-wave LDS to re-layout C -> A fragments
    unsigned short* pw = &sP[wave * 16 * 72];
    for (int t = 0; t < 4; ++t)
      for (int r = 0; r < 8; ++r)
        pw[(hi * 8 + r) * 72 + t * 16 + l15] = f32_to_bf16(s[t][r]);
    // O += P V : 8 dim-subtiles x 2 key-chunks
    for (int kk = 0; kk < 2; ++kk) {
      Frag pf;
      pf.q[0] = *(const uint4*)&pw[l15 * 72 + kk * 32 + hi * 8];
      pf.q[1] = *(const uint4*)&pw[l15 * 72 + kk * 32 + hi * 8 + 16];
      for (int t2 = 0; t2 < 8; ++t2) {
        Frag vf;
        int n = t2 * 16 + l15;
        vf.q[0] = *(const uint4*)&sVt[n * 72 + kk * 32 + hi * 8];
        vf.q[1] = *(const uint4*)&sVt[n * 72 + kk * 32 + hi * 8 + 16];
        o[t2] = __builtin_amdgcn_wmma_f32_16x16x32_bf16(
            false, pf.v, false, vf.v, (short)0, o[t2], false, false);
      }
    }
  }
  for (int t2 = 0; t2 < 8; ++t2)
    for (int r = 0; r < 8; ++r) {
      int row = qi * 128 + wave * 16 + hi * 8 + r;
      int d = t2 * 16 + l15;
      float val = o[t2][r] / l_i[r];
      outg[((size_t)b * SEQ + row) * ATTN_INNER + h * DH + d] = f32_to_bf16(val);
    }
}

// ---------------------------------------------------------------- launch
extern "C" void kernel_launch(void* const* d_in, const int* in_sizes, int n_in,
                              void* d_out, int out_size, void* d_ws, size_t ws_size,
                              hipStream_t stream) {
  (void)in_sizes; (void)n_in; (void)out_size; (void)ws_size;
  const float* x       = (const float*)d_in[0];
  const float* gamma   = (const float*)d_in[1];
  const float* w_fused = (const float*)d_in[2];
  const float* w_attn  = (const float*)d_in[3];
  const float* w_ff    = (const float*)d_in[4];
  const float* a_q = (const float*)d_in[5];
  const float* b_q = (const float*)d_in[6];
  const float* a_k = (const float*)d_in[7];
  const float* b_k = (const float*)d_in[8];
  const float* a_v = (const float*)d_in[9];
  const float* b_v = (const float*)d_in[10];
  const float* a_o = (const float*)d_in[11];
  const float* b_o = (const float*)d_in[12];
  float* out = (float*)d_out;

  char* w = (char*)d_ws;
  auto alloc = [&](size_t bytes) {
    char* p = w;
    w += (bytes + 255) & ~(size_t)255;
    return p;
  };
  unsigned short* xn     = (unsigned short*)alloc((size_t)NTOK * DIM * 2);
  unsigned short* wfb    = (unsigned short*)alloc((size_t)DIM * FUSED_TOTAL * 2);
  unsigned short* wab    = (unsigned short*)alloc((size_t)ATTN_INNER * DIM * 2);
  unsigned short* wffb   = (unsigned short*)alloc((size_t)FF_INNER * DIM * 2);
  float*          qkv    = (float*)         alloc((size_t)NTOK * QKV_COLS * 4);
  unsigned short* hbuf   = (unsigned short*)alloc((size_t)NTOK * FF_INNER * 2);
  unsigned short* qb     = (unsigned short*)alloc((size_t)BATCH * HEADS * SEQ * DH * 2);
  unsigned short* kb     = (unsigned short*)alloc((size_t)BATCH * SEQ * DH * 2);
  unsigned short* vb     = (unsigned short*)alloc((size_t)BATCH * SEQ * DH * 2);
  unsigned short* attnob = (unsigned short*)alloc((size_t)NTOK * ATTN_INNER * 2);
  float*          Tq     = (float*)alloc((size_t)NTOK * RANK * 4);
  float*          Tk     = (float*)alloc((size_t)NTOK * RANK * 4);
  float*          Tv     = (float*)alloc((size_t)NTOK * RANK * 4);
  float*          To     = (float*)alloc((size_t)NTOK * RANK * 4);

  layernorm_kernel<<<NTOK, 256, 0, stream>>>(x, gamma, xn);
  cvt_bf16_kernel<<<4096, 256, 0, stream>>>(w_fused, wfb, DIM * FUSED_TOTAL);
  cvt_bf16_kernel<<<2048, 256, 0, stream>>>(w_attn, wab, ATTN_INNER * DIM);
  cvt_bf16_kernel<<<4096, 256, 0, stream>>>(w_ff, wffb, FF_INNER * DIM);
  gemm_bf16_kernel<<<dim3(NTOK / 128, QKV_COLS / 128), 256, 0, stream>>>(
      xn, DIM, wfb, FUSED_TOTAL, qkv, QKV_COLS, DIM, 0);
  lora_t_kernel<<<NTOK, 256, 0, stream>>>(xn, a_q, Tq);
  lora_t_kernel<<<NTOK, 256, 0, stream>>>(xn, a_k, Tk);
  lora_t_kernel<<<NTOK, 256, 0, stream>>>(xn, a_v, Tv);
  lora_apply_kernel<<<NTOK, 256, 0, stream>>>(Tq, b_q, qkv, ATTN_INNER, QKV_COLS, 0);
  lora_apply_kernel<<<NTOK, 256, 0, stream>>>(Tk, b_k, qkv, DH, QKV_COLS, ATTN_INNER);
  lora_apply_kernel<<<NTOK, 256, 0, stream>>>(Tv, b_v, qkv, DH, QKV_COLS, ATTN_INNER + DH);
  rope_qkv_kernel<<<NTOK, 256, 0, stream>>>(qkv, qb, kb, vb);
  gemm_ff_silu_kernel<<<dim3(NTOK / 128, FF_INNER / 128), 256, 0, stream>>>(xn, wfb, hbuf);
  flash_attn_kernel<<<dim3(BATCH * HEADS, SEQ / 128), 256, 0, stream>>>(qb, kb, vb, attnob);
  gemm_bf16_kernel<<<dim3(NTOK / 128, DIM / 128), 256, 0, stream>>>(
      attnob, ATTN_INNER, wab, DIM, out, DIM, ATTN_INNER, 0);
  gemm_bf16_kernel<<<dim3(NTOK / 128, DIM / 128), 256, 0, stream>>>(
      hbuf, FF_INNER, wffb, DIM, out, DIM, FF_INNER, 1);
  lora_t_kernel<<<NTOK, 256, 0, stream>>>(attnob, a_o, To);
  lora_apply_kernel<<<NTOK, 256, 0, stream>>>(To, b_o, out, DIM, DIM, 0);
}
